// TP_Diff_Degra_V2_70317204570395
// MI455X (gfx1250) — matrix operand order repaired and checked by reading, not compiled, not run on hardware
//
#include <hip/hip_runtime.h>
#include <hip/hip_bf16.h>
#include <math.h>

typedef __attribute__((ext_vector_type(16))) _Float16 v16h;
typedef __attribute__((ext_vector_type(8)))  float    v8f;
typedef __attribute__((ext_vector_type(4)))  _Float16 v4h;
typedef __attribute__((ext_vector_type(4)))  unsigned int u32x4;
typedef __attribute__((ext_vector_type(8)))  int      i32x8;
typedef __attribute__((ext_vector_type(4)))  int      i32x4;

#define Bc 64
#define Tc 32
#define NN 15
#define FF 1536
#define DD 768
#define BT 2048   // Bc*Tc

union frag16 { uint4 q[2]; v16h h; };

__device__ __forceinline__ float softplus_f(float y) {
  return (y > 20.f) ? y : log1pf(__expf(y));
}

// ---------------------------------------------------------------------------
// TDM issue: 2D tile (32 rows x 32 f16), row stride NN*FF elements,
// LDS pad 4 dwords every 16 dwords -> 80B rows (matches 40-half padding).
// ---------------------------------------------------------------------------
__device__ __forceinline__ void tdm_issue(unsigned int lds_addr,
                                          unsigned long long ga) {
  u32x4 g0;
  g0[0] = 1u;                                   // count=1, user descriptor
  g0[1] = lds_addr;                             // lds_addr
  g0[2] = (unsigned int)ga;                     // global_addr[31:0]
  g0[3] = (unsigned int)((ga >> 32) & 0x1FFFFFFu) | (2u << 30); // addr hi, type=2
  i32x8 g1;
  g1[0] = (int)(0x10000u                        // data_size = 1 (2 bytes)
                | (1u << 20)                    // pad_enable
                | (3u << 22)                    // pad_interval: 16 dwords
                | (3u << 25));                  // pad_amount:  4 dwords
  g1[1] = (int)(32u << 16);                     // tensor_dim0 = 32 (low16)
  g1[2] = (int)(32u << 16);                     // tensor_dim1 = 32 (low16)
  g1[3] = (int)(32u << 16);                     // tile_dim0 = 32
  g1[4] = 32;                                   // tile_dim1 = 32, tile_dim2 = 0
  g1[5] = NN * FF;                              // tensor_dim0_stride = 23040
  g1[6] = 0;
  g1[7] = 0;
  i32x4 z4 = {0, 0, 0, 0};
#if defined(__clang_major__) && (__clang_major__ >= 23)
  i32x8 z8 = {0, 0, 0, 0, 0, 0, 0, 0};
  __builtin_amdgcn_tensor_load_to_lds(g0, g1, z4, z4, z8, 0);
#else
  __builtin_amdgcn_tensor_load_to_lds(g0, g1, z4, z4, 0);
#endif
}

// ---------------------------------------------------------------------------
// 0) Convert x (B,T,N,F) f32 -> f16 (one streaming pass)
// ---------------------------------------------------------------------------
__global__ __launch_bounds__(256)
void xh_kernel(const float* __restrict__ x, _Float16* __restrict__ xh) {
  size_t i = ((size_t)blockIdx.x * 256 + threadIdx.x) * 4;
  float4 v = *(const float4*)(x + i);
  v4h h = {(_Float16)v.x, (_Float16)v.y, (_Float16)v.z, (_Float16)v.w};
  *(v4h*)(xh + i) = h;
}

// ---------------------------------------------------------------------------
// 1) Transpose + convert weight pools: W(pool,n,F,D) f32 -> Wt(pool,n,D,F) f16
// ---------------------------------------------------------------------------
__global__ __launch_bounds__(256)
void wt_kernel(const float* __restrict__ wk, const float* __restrict__ wv,
               _Float16* __restrict__ Wt) {
  __shared__ float tile[32][33];
  int pn   = blockIdx.z;            // 0..29
  int pool = pn / NN, node = pn % NN;
  const float* W = (pool ? wv : wk) + (size_t)node * FF * DD;
  int dbase = blockIdx.x * 32, fbase = blockIdx.y * 32;
  int tx = threadIdx.x & 31, ty = threadIdx.x >> 5;   // 32 x 8
#pragma unroll
  for (int i = 0; i < 4; i++) {
    int f = fbase + ty + i * 8;
    tile[ty + i * 8][tx] = W[(size_t)f * DD + dbase + tx];
  }
  __syncthreads();
  _Float16* Wo = Wt + (size_t)pn * DD * FF;
#pragma unroll
  for (int i = 0; i < 4; i++) {
    int d = dbase + ty + i * 8;
    Wo[(size_t)d * FF + fbase + tx] = (_Float16)tile[tx][ty + i * 8];
  }
}

// ---------------------------------------------------------------------------
// 2) Label preprocessing: diff norms, global min/max, map to [-1,1]
// ---------------------------------------------------------------------------
__global__ void label_kernel(const float* __restrict__ labels,
                             float* __restrict__ nrm, float* __restrict__ mapped) {
  __shared__ float smn[1024], smx[1024];
  int tid = threadIdx.x;
  const int NE = Bc * (Tc - 1) * (NN - 1);     // 64*31*14
  float mn = 3.4e38f, mx = -3.4e38f;
  for (int i = tid; i < NE; i += 1024) {
    int j  = i % 14;
    int tq = (i / 14) % 31;
    int b  = i / (14 * 31);
    size_t o1 = (((size_t)b * Tc + tq + 1) * 14 + j) * 3;
    size_t o0 = (((size_t)b * Tc + tq) * 14 + j) * 3;
    float dx = labels[o1]     - labels[o0];
    float dy = labels[o1 + 1] - labels[o0 + 1];
    float dz = labels[o1 + 2] - labels[o0 + 2];
    float r = sqrtf(dx * dx + dy * dy + dz * dz);
    nrm[i] = r;
    mn = fminf(mn, r); mx = fmaxf(mx, r);
  }
  smn[tid] = mn; smx[tid] = mx;
  __syncthreads();
  for (int s = 512; s > 0; s >>= 1) {
    if (tid < s) {
      smn[tid] = fminf(smn[tid], smn[tid + s]);
      smx[tid] = fmaxf(smx[tid], smx[tid + s]);
    }
    __syncthreads();
  }
  mn = smn[0]; mx = smx[0];
  float scale = 2.f / (mx - mn);
  const int NM = BT * 14;
  for (int i = tid; i < NM; i += 1024) {
    int j = i % 14;
    int t = (i / 14) % Tc;
    int b = i / (14 * Tc);
    float v = 0.f;
    if (t > 0) v = (nrm[(b * 31 + (t - 1)) * 14 + j] - mn) * scale - 1.f;
    mapped[i] = v;
  }
}

// ---------------------------------------------------------------------------
// 3) Input gate: wave per (bt, n): I = sigmoid(x . w_in + b + 0.1*lab)
// ---------------------------------------------------------------------------
__global__ __launch_bounds__(256)
void gate_kernel(const float* __restrict__ x, const float* __restrict__ w_in,
                 const float* __restrict__ b_in, const float* __restrict__ mapped,
                 float* __restrict__ Ig) {
  int wv = threadIdx.x >> 5, lane = threadIdx.x & 31;
  int idx = blockIdx.x * 8 + wv;              // < 30720
  int bt = idx / NN, n = idx % NN;
  const float* xr = x + ((size_t)bt * NN + n) * FF;
  float s = 0.f;
  for (int i = lane; i < FF; i += 32) s += xr[i] * w_in[i];
#pragma unroll
  for (int m = 16; m >= 1; m >>= 1) s += __shfl_xor(s, m, 32);
  if (lane == 0) {
    float a = s + b_in[0];
    if (n < NN - 1) a += 0.1f * mapped[bt * 14 + n];
    Ig[idx] = 1.f / (1.f + __expf(-a));
  }
}

// ---------------------------------------------------------------------------
// 4) Fused WMMA GEMM + bias + LayerNorm + softplus.
//    A tiles staged by the Tensor Data Mover (double-buffered LDS),
//    B fragments streamed from L2-resident transposed f16 weights.
// ---------------------------------------------------------------------------
__global__ __launch_bounds__(256)
void gemm_keyval(const _Float16* __restrict__ xh, const _Float16* __restrict__ Wt,
                 const float* __restrict__ bias_k, const float* __restrict__ bias_v,
                 const float* __restrict__ kg, const float* __restrict__ kb,
                 const float* __restrict__ vg, const float* __restrict__ vb,
                 _Float16* __restrict__ outKV) {
  __shared__ __align__(16) _Float16 Abuf[2][32 * 40];  // 80B rows via TDM pad
  __shared__ float wsum[8][32];
  __shared__ float wsqs[8][32];
  __shared__ float rstats[32][2];

  int wg   = blockIdx.x;
  int mb   = wg & 63;
  int rest = wg >> 6;
  int node = rest % NN;
  int pool = rest / NN;

  int tid = threadIdx.x, wv = tid >> 5, lane = tid & 31;
  int hl = lane >> 4, l15 = lane & 15;
  int rbase = mb * 32;
  int wcol0 = wv * 96;

  const float* bias = pool ? bias_v : bias_k;
  const float* lng  = pool ? vg : kg;
  const float* lnb  = pool ? vb : kb;
  const _Float16* Wn = Wt + (size_t)(pool * NN + node) * DD * FF;

  unsigned int ldsA[2];
  ldsA[0] = (unsigned int)(unsigned long long)(&Abuf[0][0]);
  ldsA[1] = (unsigned int)(unsigned long long)(&Abuf[1][0]);
  unsigned long long gaBase =
      (unsigned long long)(const void*)(xh + ((size_t)rbase * NN + node) * FF);

  bool leader = (wv == 0);

  v8f acc[2][6];
  v8f vz = {0.f, 0.f, 0.f, 0.f, 0.f, 0.f, 0.f, 0.f};
#pragma unroll
  for (int i = 0; i < 2; i++)
#pragma unroll
    for (int j = 0; j < 6; j++) acc[i][j] = vz;

  if (leader) tdm_issue(ldsA[0], gaBase);   // tile 0 -> buf 0

  int cur = 0;
  const int NIT = FF / 32;                  // 48
  for (int it = 0; it < NIT; ++it) {
    if (it + 1 < NIT) {
      if (leader) {
        tdm_issue(ldsA[cur ^ 1], gaBase + (unsigned long long)(it + 1) * 64ull);
        __builtin_amdgcn_s_wait_tensorcnt(1);   // tile 'it' resident
      }
    } else if (leader) {
      __builtin_amdgcn_s_wait_tensorcnt(0);
    }
    __syncthreads();

    // A fragments from LDS (two 16B ds loads per m-tile)
    const _Float16* At = &Abuf[cur][0];
    frag16 a[2];
#pragma unroll
    for (int mt = 0; mt < 2; mt++) {
      int row = mt * 16 + l15;
      int f0  = hl ? 8 : 0;
      a[mt].q[0] = *(const uint4*)&At[row * 40 + f0];
      a[mt].q[1] = *(const uint4*)&At[row * 40 + f0 + 16];
    }
    // B fragments straight from L2-resident transposed weights + WMMA
    int k0 = it * 32;
#pragma unroll
    for (int jb = 0; jb < 6; jb++) {
      int col = wcol0 + jb * 16 + l15;
      const _Float16* bpf = Wn + (size_t)col * FF + k0 + hl * 8;
      const uint4* bp = (const uint4*)bpf;
      frag16 bfr; bfr.q[0] = bp[0]; bfr.q[1] = bp[2];
      __builtin_prefetch(bpf + 32, 0, 1);   // next K tile -> global_prefetch
      acc[0][jb] = __builtin_amdgcn_wmma_f32_16x16x32_f16(
          false, a[0].h, false, bfr.h, (short)0, acc[0][jb], false, false);
      acc[1][jb] = __builtin_amdgcn_wmma_f32_16x16x32_f16(
          false, a[1].h, false, bfr.h, (short)0, acc[1][jb], false, false);
    }
    __syncthreads();   // all reads of Abuf[cur] done before it is re-filled
    cur ^= 1;
  }

  // Epilogue: + bias, LayerNorm over the full 768-wide row, softplus, store f16
  float g6[6], be6[6], bi6[6];
  int d6[6];
#pragma unroll
  for (int jb = 0; jb < 6; jb++) {
    int d = wcol0 + jb * 16 + l15;
    d6[jb]  = d;
    bi6[jb] = bias[node * DD + d];
    g6[jb]  = lng[d];
    be6[jb] = lnb[d];
  }
#pragma unroll
  for (int mt = 0; mt < 2; mt++)
#pragma unroll
    for (int jb = 0; jb < 6; jb++)
#pragma unroll
      for (int g = 0; g < 8; g++) acc[mt][jb][g] += bi6[jb];

  // per-row partial sums across this wave's 96 columns
#pragma unroll
  for (int mt = 0; mt < 2; mt++) {
#pragma unroll
    for (int g = 0; g < 8; g++) {
      float s = 0.f, q = 0.f;
#pragma unroll
      for (int jb = 0; jb < 6; jb++) {
        float t = acc[mt][jb][g];
        s += t; q += t * t;
      }
#pragma unroll
      for (int m = 1; m < 16; m <<= 1) {
        s += __shfl_xor(s, m, 32);
        q += __shfl_xor(q, m, 32);
      }
      if (l15 == 0) {
        int row = mt * 16 + g + hl * 8;
        wsum[wv][row] = s;
        wsqs[wv][row] = q;
      }
    }
  }
  __syncthreads();
  if (tid < 32) {
    float s = 0.f, q = 0.f;
#pragma unroll
    for (int w = 0; w < 8; w++) { s += wsum[w][tid]; q += wsqs[w][tid]; }
    float mean = s * (1.f / 768.f);
    float var  = q * (1.f / 768.f) - mean * mean;
    rstats[tid][0] = mean;
    rstats[tid][1] = rsqrtf(var + 1e-5f);
  }
  __syncthreads();

  const float inv_sqrt_d = 0.036084391824351615f;   // 1/sqrt(768)
  _Float16* outp = outKV + (size_t)pool * BT * NN * DD;
#pragma unroll
  for (int mt = 0; mt < 2; mt++) {
#pragma unroll
    for (int g = 0; g < 8; g++) {
      int row = mt * 16 + g + hl * 8;
      float mean = rstats[row][0], rsig = rstats[row][1];
#pragma unroll
      for (int jb = 0; jb < 6; jb++) {
        float y  = (acc[mt][jb][g] - mean) * rsig * g6[jb] + be6[jb];
        float sp = softplus_f(y);
        if (pool == 0) sp *= inv_sqrt_d;
        outp[((size_t)(rbase + row) * NN + node) * DD + d6[jb]] = (_Float16)sp;
      }
    }
  }
}

// ---------------------------------------------------------------------------
// 5) now = relu(LN_m(key . val^T)): one wave per (b,t), 16x16x768 via WMMA
// ---------------------------------------------------------------------------
__global__ __launch_bounds__(256)
void now_kernel(const _Float16* __restrict__ key, const _Float16* __restrict__ val,
                const float* __restrict__ ng, const float* __restrict__ nb,
                float* __restrict__ now) {
  int wv = threadIdx.x >> 5, lane = threadIdx.x & 31;
  int bt = blockIdx.x * 8 + wv;               // < 2048
  int hl = lane >> 4, l15 = lane & 15;
  int arow = (l15 > 14) ? 14 : l15;           // clamp padded row/col in-bounds
  const _Float16* kbase = key + (size_t)bt * NN * DD;
  const _Float16* vbase = val + (size_t)bt * NN * DD;

  v8f acc = {0.f, 0.f, 0.f, 0.f, 0.f, 0.f, 0.f, 0.f};
  int f0 = hl ? 8 : 0;
  for (int k0 = 0; k0 < DD; k0 += 32) {
    frag16 a, b;
    const uint4* ap = (const uint4*)(kbase + (size_t)arow * DD + k0 + f0);
    a.q[0] = ap[0];
    a.q[1] = ap[2];                           // +32 bytes = K+16
    const uint4* bp = (const uint4*)(vbase + (size_t)arow * DD + k0 + hl * 8);
    b.q[0] = bp[0];
    b.q[1] = bp[2];
    acc = __builtin_amdgcn_wmma_f32_16x16x32_f16(
        false, a.h, false, b.h, (short)0, acc, false, false);
  }

  bool colok = (l15 < NN);
  float gm = ng[arow], bm = nb[arow];
  const float inv15 = 1.f / 15.f;
#pragma unroll
  for (int g = 0; g < 8; g++) {
    int n = g + hl * 8;
    float v  = acc[g];
    float vv = colok ? v : 0.f;
    float s = vv, q = vv * vv;
#pragma unroll
    for (int m = 1; m < 16; m <<= 1) {
      s += __shfl_xor(s, m, 32);
      q += __shfl_xor(q, m, 32);
    }
    float mean = s * inv15;
    float var  = q * inv15 - mean * mean;
    float y = (v - mean) * rsqrtf(var + 1e-5f) * gm + bm;
    y = y > 0.f ? y : 0.f;
    if (colok && n < NN)
      now[((size_t)bt * NN + n) * NN + l15] = y;
  }
}

// ---------------------------------------------------------------------------
// 6) Sequential scan over T: cell = (1-I)*cell + I*now
// ---------------------------------------------------------------------------
__global__ void scan_kernel(const float* __restrict__ cell_past,
                            const float* __restrict__ Ig,
                            const float* __restrict__ now,
                            float* __restrict__ out) {
  int i = blockIdx.x * 256 + threadIdx.x;
  if (i >= Bc * NN * NN) return;
  int m = i % NN, n = (i / NN) % NN, b = i / (NN * NN);
  float c = cell_past[i];
  for (int t = 0; t < Tc; t++) {
    int bt = b * Tc + t;
    float I = Ig[bt * NN + n];
    c = (1.f - I) * c + I * now[((size_t)bt * NN + n) * NN + m];
  }
  out[i] = c;
}

// ---------------------------------------------------------------------------
extern "C" void kernel_launch(void* const* d_in, const int* in_sizes, int n_in,
                              void* d_out, int out_size, void* d_ws, size_t ws_size,
                              hipStream_t stream) {
  const float* x         = (const float*)d_in[0];
  const float* labels    = (const float*)d_in[1];
  const float* cell_past = (const float*)d_in[2];
  const float* wk        = (const float*)d_in[3];
  const float* wvp       = (const float*)d_in[4];
  const float* bias_k    = (const float*)d_in[5];
  const float* bias_v    = (const float*)d_in[6];
  const float* w_in      = (const float*)d_in[7];
  const float* b_in      = (const float*)d_in[8];
  const float* kg        = (const float*)d_in[9];
  const float* kb        = (const float*)d_in[10];
  const float* vg        = (const float*)d_in[11];
  const float* vb        = (const float*)d_in[12];
  const float* ng        = (const float*)d_in[13];
  const float* nb        = (const float*)d_in[14];
  (void)in_sizes; (void)n_in; (void)out_size; (void)ws_size;

  // workspace carve (bytes, 16B aligned)
  char* ws = (char*)d_ws;
  const size_t szWt  = (size_t)2 * NN * DD * FF * 2;      // 70,778,880
  const size_t szXh  = (size_t)BT * NN * FF * 2;          // 94,371,840
  const size_t szKV  = (size_t)2 * BT * NN * DD * 2;      // 94,371,840
  const size_t szNow = (size_t)BT * NN * NN * 4;          //  1,843,200
  const size_t szIg  = (size_t)BT * NN * 4;               //    122,880
  const size_t szMap = (size_t)BT * 14 * 4;               //    114,688

  _Float16* Wt   = (_Float16*)ws;
  _Float16* Xh   = (_Float16*)(ws + szWt);
  _Float16* KV   = (_Float16*)(ws + szWt + szXh);
  float* nowbuf  = (float*)(ws + szWt + szXh + szKV);
  float* Ig      = (float*)(ws + szWt + szXh + szKV + szNow);
  float* mapped  = (float*)(ws + szWt + szXh + szKV + szNow + szIg);
  float* nrm     = (float*)(ws + szWt + szXh + szKV + szNow + szIg + szMap);

  _Float16* keyp = KV;
  _Float16* valp = KV + (size_t)BT * NN * DD;

  xh_kernel<<<(unsigned)((size_t)BT * NN * FF / 4 / 256), 256, 0, stream>>>(x, Xh);
  wt_kernel<<<dim3(DD / 32, FF / 32, 30), 256, 0, stream>>>(wk, wvp, Wt);
  label_kernel<<<1, 1024, 0, stream>>>(labels, nrm, mapped);
  gate_kernel<<<(BT * NN) / 8, 256, 0, stream>>>(x, w_in, b_in, mapped, Ig);
  gemm_keyval<<<64 * NN * 2, 256, 0, stream>>>(Xh, Wt, bias_k, bias_v,
                                               kg, kb, vg, vb, KV);
  now_kernel<<<BT / 8, 256, 0, stream>>>(keyp, valp, ng, nb, nowbuf);
  scan_kernel<<<(Bc * NN * NN + 255) / 256, 256, 0, stream>>>(
      cell_past, Ig, nowbuf, (float*)d_out);
}
